// Sparse_3d_Convolution_Block_69526930588487
// MI455X (gfx1250) — matrix operand order repaired and checked by reading, not compiled, not run on hardware
//
#include <hip/hip_runtime.h>
#include <stdint.h>

#define N_PTS    180000
#define CIN      64
#define COUT     64
#define NK       27
#define TILES    (N_PTS / 16)      // 11250, exact (N divisible by 16)
#define CONV_BLOCKS 352            // 8 waves * 4 subtiles * 16 rows = 512 rows/block
#define RED_ROWS 512
#define RED_BLOCKS 352

typedef __attribute__((ext_vector_type(16))) __bf16 v16bf;
typedef __attribute__((ext_vector_type(8)))  float  v8f;

union BfPack { uint4 q[2]; v16bf v; };

__device__ __forceinline__ unsigned short f2bf(float x) {
  unsigned u = __float_as_uint(x);
  unsigned r = u + 0x7FFFu + ((u >> 16) & 1u);   // round-to-nearest-even
  return (unsigned short)(r >> 16);
}

// ---------------------------------------------------------------------------
// Weight prep: f32 weight[27][64][64] -> bf16 hi/lo, pre-swizzled into the
// B-operand lane layout this kernel uses (contiguous k, same as A):
//   for (k, kb=K/32, ct=col/16):  word[(lane)*8 + v]
//   lane = (col&15) + 16*((K%32)>=16), within-half k' = K%16, v = k'/2,
//   low/high u16 half = k' even/odd.
// ---------------------------------------------------------------------------
__global__ __launch_bounds__(256) void prep_w_k(const float* __restrict__ w,
    unsigned short* __restrict__ bh, unsigned short* __restrict__ bl) {
  int i = blockIdx.x * 256 + threadIdx.x;
  if (i >= NK * CIN * COUT) return;
  int k   = i >> 12;       // /4096
  int rem = i & 4095;
  int K   = rem >> 6;      // cin index
  int col = rem & 63;      // cout index
  float x = w[i];
  unsigned short h = f2bf(x);
  float hf = __uint_as_float(((unsigned)h) << 16);
  unsigned short l = f2bf(x - hf);
  int kb   = K >> 5;
  int Kr   = K & 31;
  int lane = (col & 15) + ((Kr & 16) ? 16 : 0);
  int kk   = Kr & 15;
  int ct   = col >> 4;
  size_t word = ((size_t)((k * 2 + kb) * 4 + ct) * 32 + lane) * 8 + (kk >> 1);
  size_t idx  = word * 2 + (kk & 1);
  bh[idx] = h;
  bl[idx] = l;
}

// feats f32 -> bf16 hi + bf16 lo (row-major [N+1][64]; row N is the zero row
// used as the target of "no neighbor" gathers, so the hot loop needs no
// predication at all).
__global__ __launch_bounds__(256) void prep_feats_k(const float* __restrict__ f,
    unsigned short* __restrict__ fh, unsigned short* __restrict__ fl) {
  int i = blockIdx.x * 256 + threadIdx.x;
  if (i >= (N_PTS + 1) * CIN) return;
  float x = (i < N_PTS * CIN) ? f[i] : 0.0f;
  unsigned short h = f2bf(x);
  float hf = __uint_as_float(((unsigned)h) << 16);
  unsigned short l = f2bf(x - hf);
  fh[i] = h;
  fl[i] = l;
}

// nbr[k][i] = i for center (k==13), else N_PTS (zero row). scatter_k fills
// real neighbors afterwards.
__global__ __launch_bounds__(256) void init_nbr_k(int* __restrict__ nbr) {
  int i = blockIdx.x * 256 + threadIdx.x;
  int k = blockIdx.y;
  if (i >= N_PTS) return;
  nbr[(size_t)k * N_PTS + i] = (k == 13) ? i : N_PTS;
}

// Invert rulebook: pair (in=i, out=j) for offset k  ==>  nbr[k'][j] = i
__global__ __launch_bounds__(256) void scatter_k(const int* __restrict__ in_idx,
    const int* __restrict__ out_idx, int* __restrict__ nbr, int P) {
  int p = blockIdx.x * 256 + threadIdx.x;
  int k = blockIdx.y;                       // 0..25
  if (p >= P) return;
  int j = out_idx[(size_t)k * P + p];
  if (j < 0 || j >= N_PTS) return;          // padding rows have j == N
  int kp = (k < 13) ? k : (k + 1);
  nbr[(size_t)kp * N_PTS + j] = in_idx[(size_t)k * P + p];
}

// ---------------------------------------------------------------------------
// Main conv: per wave, 4 tiles of 16 output rows x 64 cols.
// acc[s][ct] (f32) += sum_k gather(feats, nbr[k]) @ w[k], computed as
//   Ah@Bh + Al@Bh + Ah@Bl  in bf16 WMMA with f32 accumulate (fp32-class acc.)
// Weights for offset k staged in double-buffered LDS (16KB/buffer).
// No divergence in the k-loop: missing neighbors gather the zero row, and the
// grid tail clamps its gather address (its accumulators are never stored).
// ---------------------------------------------------------------------------
__global__ __launch_bounds__(256) void conv_k(const int* __restrict__ nbr,
    const unsigned short* __restrict__ fh, const unsigned short* __restrict__ fl,
    const unsigned int* __restrict__ bh, const unsigned int* __restrict__ bl,
    float* __restrict__ out) {
  __shared__ unsigned int lds[2][4096];   // [buf][ hi:0..2047 | lo:2048..4095 ]
  const int tid   = threadIdx.x;
  const int wave  = tid >> 5;
  const int lane  = tid & 31;
  const int m     = lane & 15;                 // row within 16-row tile
  const int tile0 = (blockIdx.x * 8 + wave) * 4;
  const int k0    = (lane < 16) ? 0 : 16;      // A-operand K base per lane half

  // gather-address row offsets (clamped for the grid tail; stores are guarded)
  int roff[4];
#pragma unroll
  for (int s = 0; s < 4; ++s) {
    int t = tile0 + s;
    if (t > TILES - 1) t = TILES - 1;
    roff[s] = t * 16 + m;
  }

  const v8f vz = {0.f, 0.f, 0.f, 0.f, 0.f, 0.f, 0.f, 0.f};
  v8f acc[4][4];
#pragma unroll
  for (int s = 0; s < 4; ++s)
#pragma unroll
    for (int ct = 0; ct < 4; ++ct) acc[s][ct] = vz;

  // stage k=0 weights into LDS buffer 0
  {
    const uint4* ph = (const uint4*)bh;
    const uint4* pl = (const uint4*)bl;
    uint4 h0 = ph[tid * 2], h1 = ph[tid * 2 + 1];
    uint4 l0 = pl[tid * 2], l1 = pl[tid * 2 + 1];
    ((uint4*)&lds[0][0])[tid * 2]     = h0;
    ((uint4*)&lds[0][0])[tid * 2 + 1] = h1;
    ((uint4*)&lds[0][2048])[tid * 2]     = l0;
    ((uint4*)&lds[0][2048])[tid * 2 + 1] = l1;
  }
  __syncthreads();

  const int* nbr_k = nbr;   // advances by N_PTS per k
  for (int k = 0; k < NK; ++k) {
    const int cb = k & 1;
    const bool pf = (k + 1 < NK);
    uint4 h0 = make_uint4(0,0,0,0), h1 = h0, l0 = h0, l1 = h0;
    if (pf) {  // prefetch next offset's weights (global -> regs)
      const uint4* ph = (const uint4*)(bh + (size_t)(k + 1) * 2048);
      const uint4* pl = (const uint4*)(bl + (size_t)(k + 1) * 2048);
      h0 = ph[tid * 2]; h1 = ph[tid * 2 + 1];
      l0 = pl[tid * 2]; l1 = pl[tid * 2 + 1];
    }

    int rr[4];
#pragma unroll
    for (int s = 0; s < 4; ++s) rr[s] = nbr_k[roff[s]];
    nbr_k += N_PTS;

#pragma unroll
    for (int kb = 0; kb < 2; ++kb) {
#pragma unroll
      for (int s = 0; s < 4; ++s) {
        BfPack ah, al;
        {
          const unsigned short* bhp = fh + (size_t)rr[s] * CIN + kb * 32 + k0;
          const unsigned short* blp = fl + (size_t)rr[s] * CIN + kb * 32 + k0;
          ah.q[0] = *(const uint4*)(bhp);
          ah.q[1] = *(const uint4*)(bhp + 8);
          al.q[0] = *(const uint4*)(blp);
          al.q[1] = *(const uint4*)(blp + 8);
        }
#pragma unroll
        for (int ct = 0; ct < 4; ++ct) {
          BfPack bhv, blv;
          const uint4* pb  = (const uint4*)&lds[cb][(kb * 4 + ct) * 256 + lane * 8];
          const uint4* pb2 = (const uint4*)&lds[cb][2048 + (kb * 4 + ct) * 256 + lane * 8];
          bhv.q[0] = pb[0];  bhv.q[1] = pb[1];
          blv.q[0] = pb2[0]; blv.q[1] = pb2[1];
          v8f c = acc[s][ct];
          c = __builtin_amdgcn_wmma_f32_16x16x32_bf16(false, ah.v, false, bhv.v,
                                                      (short)0, c, false, false);
          c = __builtin_amdgcn_wmma_f32_16x16x32_bf16(false, al.v, false, bhv.v,
                                                      (short)0, c, false, false);
          c = __builtin_amdgcn_wmma_f32_16x16x32_bf16(false, ah.v, false, blv.v,
                                                      (short)0, c, false, false);
          acc[s][ct] = c;
        }
      }
    }

    __syncthreads();            // everyone done reading lds[cb] (and lds[cb^1] long ago)
    if (pf) {
      const int nb = (k + 1) & 1;
      ((uint4*)&lds[nb][0])[tid * 2]     = h0;
      ((uint4*)&lds[nb][0])[tid * 2 + 1] = h1;
      ((uint4*)&lds[nb][2048])[tid * 2]     = l0;
      ((uint4*)&lds[nb][2048])[tid * 2 + 1] = l1;
    }
    __syncthreads();            // staged weights visible before next compute
  }

  // C/D layout: VGPR j, lanes 0-15 -> row j, lanes 16-31 -> row j+8; col = lane&15
#pragma unroll
  for (int s = 0; s < 4; ++s) {
    int t = tile0 + s;
    if (t >= TILES) continue;
#pragma unroll
    for (int ct = 0; ct < 4; ++ct) {
#pragma unroll
      for (int j = 0; j < 8; ++j) {
        int mr = (lane < 16) ? j : (j + 8);
        out[(size_t)(t * 16 + mr) * COUT + ct * 16 + (lane & 15)] = acc[s][ct][j];
      }
    }
  }
}

// ---------------------------------------------------------------------------
// Deterministic two-stage per-channel sum / sum-of-squares reduction
// ---------------------------------------------------------------------------
__global__ __launch_bounds__(256) void reduce1_k(const float* __restrict__ x,
                                                 float* __restrict__ partial) {
  int c = threadIdx.x & 63;
  int g = threadIdx.x >> 6;
  int r0 = blockIdx.x * RED_ROWS;
  int r1 = (r0 + RED_ROWS < N_PTS) ? (r0 + RED_ROWS) : N_PTS;
  float s = 0.f, ss = 0.f;
  for (int r = r0 + g; r < r1; r += 4) {
    float v = x[(size_t)r * COUT + c];
    s += v;
    ss += v * v;
  }
  __shared__ float sm[2][4][64];
  sm[0][g][c] = s;
  sm[1][g][c] = ss;
  __syncthreads();
  if (g == 0) {
    s  = sm[0][0][c] + sm[0][1][c] + sm[0][2][c] + sm[0][3][c];
    ss = sm[1][0][c] + sm[1][1][c] + sm[1][2][c] + sm[1][3][c];
    partial[(size_t)blockIdx.x * 128 + c]      = s;
    partial[(size_t)blockIdx.x * 128 + 64 + c] = ss;
  }
}

__global__ __launch_bounds__(128) void reduce2_k(const float* __restrict__ partial,
                                                 float* __restrict__ stats) {
  int t = threadIdx.x;  // 0..127 : [0..63]=sum, [64..127]=sumsq
  float a = 0.f;
  for (int b = 0; b < RED_BLOCKS; ++b) a += partial[(size_t)b * 128 + t];
  stats[t] = a;
}

__global__ __launch_bounds__(256) void bn_k(float* __restrict__ out,
    const float* __restrict__ stats, const float* __restrict__ gamma,
    const float* __restrict__ beta) {
  int i = blockIdx.x * 256 + threadIdx.x;
  if (i >= N_PTS * COUT) return;
  int c = i & 63;
  float mean = stats[c] * (1.0f / N_PTS);
  float var  = stats[64 + c] * (1.0f / N_PTS) - mean * mean;
  float inv  = rsqrtf(var + 1e-5f);
  float y = (out[i] - mean) * inv * gamma[c] + beta[c];
  out[i] = fmaxf(y, 0.0f);
}

// ---------------------------------------------------------------------------
extern "C" void kernel_launch(void* const* d_in, const int* in_sizes, int n_in,
                              void* d_out, int out_size, void* d_ws, size_t ws_size,
                              hipStream_t stream) {
  const float* feats   = (const float*)d_in[0];
  const float* weight  = (const float*)d_in[1];
  const float* gamma   = (const float*)d_in[2];
  const float* beta    = (const float*)d_in[3];
  const int*   in_idx  = (const int*)d_in[4];
  const int*   out_idx = (const int*)d_in[5];
  const int P = in_sizes[4] / 26;

  char* wsp = (char*)d_ws;
  auto take = [&](size_t bytes) -> char* {
    char* p = wsp;
    wsp += (bytes + 255) & ~(size_t)255;
    return p;
  };
  int*            nbr   = (int*)take((size_t)NK * N_PTS * 4);                  // 19.44 MB
  unsigned short* fh    = (unsigned short*)take((size_t)(N_PTS + 1) * CIN * 2);// 23.04 MB
  unsigned short* fl    = (unsigned short*)take((size_t)(N_PTS + 1) * CIN * 2);// 23.04 MB
  unsigned int*   bhp   = (unsigned int*)take((size_t)NK * 2048 * 4);          // 216 KB
  unsigned int*   blp   = (unsigned int*)take((size_t)NK * 2048 * 4);          // 216 KB
  float*          part  = (float*)take((size_t)RED_BLOCKS * 128 * 4);
  float*          stats = (float*)take(128 * 4);
  float* out = (float*)d_out;

  prep_w_k<<<(NK * CIN * COUT + 255) / 256, 256, 0, stream>>>(
      weight, (unsigned short*)bhp, (unsigned short*)blp);
  prep_feats_k<<<((N_PTS + 1) * CIN + 255) / 256, 256, 0, stream>>>(feats, fh, fl);
  init_nbr_k<<<dim3((N_PTS + 255) / 256, NK), 256, 0, stream>>>(nbr);
  scatter_k<<<dim3((P + 255) / 256, 26), 256, 0, stream>>>(in_idx, out_idx, nbr, P);
  conv_k<<<CONV_BLOCKS, 256, 0, stream>>>(nbr, fh, fl, bhp, blp, out);
  reduce1_k<<<RED_BLOCKS, 256, 0, stream>>>(out, part);
  reduce2_k<<<1, 128, 0, stream>>>(part, stats);
  bn_k<<<(N_PTS * COUT + 255) / 256, 256, 0, stream>>>(out, stats, gamma, beta);
}
